// Cross_Scale_Mamba_Block_84138409328978
// MI455X (gfx1250) — matrix operand order repaired and checked by reading, not compiled, not run on hardware
//
#include <hip/hip_runtime.h>
#include <hip/hip_bf16.h>

#define B_    8
#define H_    64
#define W_    64
#define L_    4096
#define CC    32
#define DI_   64
#define NN    16
#define KK    4
#define EPSF  1e-5f

typedef __attribute__((ext_vector_type(16))) _Float16 v16h;
typedef __attribute__((ext_vector_type(8)))  float    v8f;
typedef __attribute__((ext_vector_type(2)))  _Float16 h2;

// ---------------- WMMA fragment loaders (wave32, gfx1250 layouts) ----------------
// A-matrix 16x32 f16: lane t holds row (t&15); VGPR j<4: K=2j + 8*(t>>4); j>=4: K=16+2(j-4)+8*(t>>4)
__device__ __forceinline__ v16h a_frag_f16(const _Float16* row, int lane, int kofs) {
  v16h a; const int h8 = (lane >> 4) * 8;
#pragma unroll
  for (int j = 0; j < 8; ++j) {
    int k = ((j < 4) ? (2 * j) : (16 + 2 * (j - 4))) + h8 + kofs;
    h2 p = *(const h2*)(row + k);
    a[2 * j] = p.x; a[2 * j + 1] = p.y;
  }
  return a;
}
__device__ __forceinline__ v16h a_frag_f32(const float* row, int lane, int kofs) {
  v16h a; const int h8 = (lane >> 4) * 8;
#pragma unroll
  for (int j = 0; j < 8; ++j) {
    int k = ((j < 4) ? (2 * j) : (16 + 2 * (j - 4))) + h8 + kofs;
    float2 p = *(const float2*)(row + k);
    a[2 * j] = (_Float16)p.x; a[2 * j + 1] = (_Float16)p.y;
  }
  return a;
}
// B-matrix 32x16 f16 (pre-transposed storage BT[n][k]): lane t holds col (t&15); VGPR j<4: K=2j + 8*(t>>4);
// j>=4: K=16+2(j-4)+8*(t>>4)
__device__ __forceinline__ v16h b_frag_f16(const _Float16* row, int lane, int kofs) {
  v16h b; const int kb = (lane >> 4) * 8;
#pragma unroll
  for (int j = 0; j < 8; ++j) {
    int k = ((j < 4) ? (2 * j) : (16 + 2 * (j - 4))) + kb + kofs;
    h2 p = *(const h2*)(row + k);
    b[2 * j] = p.x; b[2 * j + 1] = p.y;
  }
  return b;
}

__device__ __forceinline__ float siluf(float x) { return x / (1.f + __expf(-x)); }
__device__ __forceinline__ float softplusf(float x) {
  return (x > 20.f) ? x : __logf(1.f + __expf(x));
}

// ---------------- weight f16 conversion (once) ----------------
__global__ void k_cvt_weights(const float* inw, const float* xpw, const float* opw,
                              _Float16* w16in, _Float16* w16xp, _Float16* w16op) {
  int t = blockIdx.x * blockDim.x + threadIdx.x;
  if (t < 128 * 32) w16in[t] = (_Float16)inw[t];
  if (t < 32 * 64)  w16op[t] = (_Float16)opw[t];
  if (t < KK * 48 * 64) {
    int k = t / (48 * 64); int r = t % (48 * 64); int c = r / 64; int d = r % 64;
    w16xp[t] = (c < 34) ? (_Float16)xpw[(k * 34 + c) * 64 + d] : (_Float16)0.f;
  }
}

// ---------------- axial depthwise convs ----------------
__global__ void k_dw_mw(const float* x, const float* w, const float* bias,
                        float* out, int cbase, int dil) {
  int idx = blockIdx.x * blockDim.x + threadIdx.x;   // (b*32+c)*L + l
  int l = idx & 4095; int bc = idx >> 12; int c = bc & 31; int b = bc >> 5;
  int hh = l >> 6, ww = l & 63;
  const float* xin = x + (size_t)(b * 128 + cbase + c) * L_;
  float acc = bias[c];
#pragma unroll
  for (int j = 0; j < 7; ++j) {
    int wj = ww + (j - 3) * dil;
    if (wj >= 0 && wj < 64) acc += w[c * 7 + j] * xin[hh * 64 + wj];
  }
  out[idx] = acc;
}

__global__ void k_dw_mh(const float* in, const float* w, const float* bias,
                        float* out, int dil) {
  int idx = blockIdx.x * blockDim.x + threadIdx.x;
  int l = idx & 4095; int bc = idx >> 12; int c = bc & 31;
  int hh = l >> 6, ww = l & 63;
  const float* p = in + (size_t)bc * L_;
  float acc = bias[c];
#pragma unroll
  for (int j = 0; j < 7; ++j) {
    int hj = hh + (j - 3) * dil;
    if (hj >= 0 && hj < 64) acc += w[c * 7 + j] * p[hj * 64 + ww];
  }
  out[idx] = acc;
}

__global__ void k_dw_c(const float* in, const float* x, const float* w, const float* bias,
                       float* tout, int cbase, int dil) {
  int idx = blockIdx.x * blockDim.x + threadIdx.x;
  int l = idx & 4095; int bc = idx >> 12; int c = bc & 31; int b = bc >> 5;
  int hh = l >> 6, ww = l & 63;
  const float* p = in + (size_t)bc * L_;
  float acc = bias[c];
#pragma unroll
  for (int i = 0; i < 3; ++i)
#pragma unroll
    for (int j = 0; j < 3; ++j) {
      int hj = hh + (i - 1) * dil, wj = ww + (j - 1) * dil;
      if (hj >= 0 && hj < 64 && wj >= 0 && wj < 64)
        acc += w[c * 9 + i * 3 + j] * p[hj * 64 + wj];
    }
  acc += x[(size_t)(b * 128 + cbase + c) * L_ + l];           // skip
  tout[((size_t)b * L_ + l) * CC + c] = acc;                  // NHWC
}

// ---------------- layernorm(32) -> f16 ----------------
__global__ void k_ln(const float* t, const float* lnw, const float* lnb, _Float16* ln16) {
  int lane = threadIdx.x & 31;
  int row = blockIdx.x * 8 + (threadIdx.x >> 5);              // b*L + l
  float v = t[(size_t)row * CC + lane];
  float s = v, sq = v * v;
#pragma unroll
  for (int m = 16; m >= 1; m >>= 1) { s += __shfl_xor(s, m, 32); sq += __shfl_xor(sq, m, 32); }
  float mu = s * (1.f / 32.f);
  float var = sq * (1.f / 32.f) - mu * mu;
  float nv = (v - mu) * rsqrtf(var + EPSF) * lnw[lane] + lnb[lane];
  ln16[(size_t)row * CC + lane] = (_Float16)nv;
}

// ---------------- in_proj: (32768x32) x (32x128) via WMMA ----------------
__global__ void k_gemm_inproj(const _Float16* A, const _Float16* Bw, float* xz) {
  int lane = threadIdx.x & 31;
  int tile = blockIdx.x * 8 + (threadIdx.x >> 5);             // 16384 tiles
  int m0 = (tile >> 3) * 16, n0 = (tile & 7) * 16;
  int r15 = lane & 15;
  v16h a = a_frag_f16(A + (size_t)(m0 + r15) * 32, lane, 0);
  v16h b = b_frag_f16(Bw + (size_t)(n0 + r15) * 32, lane, 0);
  v8f c = {0.f, 0.f, 0.f, 0.f, 0.f, 0.f, 0.f, 0.f};
  c = __builtin_amdgcn_wmma_f32_16x16x32_f16(false, a, false, b, (short)0, c, false, false);
  int moff = (lane >> 4) * 8;
  float* outp = xz + (size_t)(m0 + moff) * 128 + n0 + r15;
#pragma unroll
  for (int r = 0; r < 8; ++r) outp[(size_t)r * 128] = c[r];
}

// ---------------- depthwise 3x3 on xc + silu ----------------
__global__ void k_conv_silu(const float* xz, const float* cw, const float* cb, float* xcv) {
  int idx = blockIdx.x * blockDim.x + threadIdx.x;            // (b*L+l)*64 + d
  int d = idx & 63; int bl = idx >> 6; int l = bl & 4095; int b = bl >> 12;
  int hh = l >> 6, ww = l & 63;
  float acc = cb[d];
#pragma unroll
  for (int i = 0; i < 3; ++i)
#pragma unroll
    for (int j = 0; j < 3; ++j) {
      int hj = hh + i - 1, wj = ww + j - 1;
      if (hj >= 0 && hj < 64 && wj >= 0 && wj < 64)
        acc += cw[d * 9 + i * 3 + j] * xz[((size_t)b * L_ + hj * 64 + wj) * 128 + d];
    }
  xcv[idx] = siluf(acc);
}

// ---------------- build 4 scan directions ----------------
__global__ void k_build_xs(const float* xcv, float* xs) {
  int idx = blockIdx.x * blockDim.x + threadIdx.x;
  int d = idx & 63; int bl = idx >> 6; int l = bl & 4095; int b = bl >> 12;
  float v = xcv[idx];
  int hh = l >> 6, ww = l & 63;
  int lw = ww * 64 + hh;
  size_t base = (size_t)b * 4 * L_ * 64;
  xs[base + ((size_t)0 * L_ + l)          * 64 + d] = v;
  xs[base + ((size_t)1 * L_ + lw)         * 64 + d] = v;
  xs[base + ((size_t)2 * L_ + (4095 - l)) * 64 + d] = v;
  xs[base + ((size_t)3 * L_ + (4095 - lw))* 64 + d] = v;
}

// ---------------- x_proj: per (b,k): (4096x64) x (64x48) via WMMA ----------------
__global__ void k_gemm_xproj(const float* xs, const _Float16* Bw, float* xdbl) {
  int lane = threadIdx.x & 31;
  int tile = blockIdx.x * 8 + (threadIdx.x >> 5);             // 24576 tiles
  int bk = tile / 768; int rem = tile % 768;
  int m0 = (rem / 3) * 16, n0 = (rem % 3) * 16;
  int r15 = lane & 15;
  const float*    Arow = xs + ((size_t)bk * L_ + m0 + r15) * 64;
  const _Float16* Brow = Bw + ((size_t)(bk & 3) * 48 + n0 + r15) * 64;
  v8f c = {0.f, 0.f, 0.f, 0.f, 0.f, 0.f, 0.f, 0.f};
#pragma unroll
  for (int s = 0; s < 2; ++s) {
    v16h a = a_frag_f32(Arow, lane, 32 * s);
    v16h b = b_frag_f16(Brow, lane, 32 * s);
    c = __builtin_amdgcn_wmma_f32_16x16x32_f16(false, a, false, b, (short)0, c, false, false);
  }
  int moff = (lane >> 4) * 8;
  float* outp = xdbl + ((size_t)bk * L_ + m0 + moff) * 48 + n0 + r15;
#pragma unroll
  for (int r = 0; r < 8; ++r) outp[(size_t)r * 48] = c[r];
}

// ---------------- selective scan (in-place over xs): y + D*u ----------------
__global__ void k_scan(const float* xdbl, float* xs, const float* dtw, const float* dtb,
                       const float* Alog, const float* Dp) {
  int bk = blockIdx.x;                                        // 0..31
  int d = threadIdx.x;                                        // 0..63
  int k = bk & 3; int kd = k * 64 + d;
  float a[NN], h[NN];
#pragma unroll
  for (int n = 0; n < NN; ++n) { a[n] = -__expf(Alog[kd * NN + n]); h[n] = 0.f; }
  float w0 = dtw[kd * 2 + 0], w1 = dtw[kd * 2 + 1];
  float bcst = dtb[kd], Dk = Dp[kd];
  const float* dbl = xdbl + (size_t)bk * L_ * 48;
  float* u = xs + (size_t)bk * L_ * 64;
  for (int l = 0; l < L_; ++l) {
    const float* row = dbl + (size_t)l * 48;
    float delta = softplusf(bcst + w0 * row[0] + w1 * row[1]);
    float uv = u[(size_t)l * 64 + d];
    float du = delta * uv;
    float y = 0.f;
#pragma unroll
    for (int n = 0; n < NN; ++n) {
      h[n] = h[n] * __expf(delta * a[n]) + du * row[2 + n];
      y += h[n] * row[18 + n];
    }
    u[(size_t)l * 64 + d] = y + Dk * uv;
  }
}

// ---------------- combine 4 dirs + LN(64) + silu(z) gate -> f16 ----------------
__global__ void k_combine(const float* oy, const float* xz, const float* onw, const float* onb,
                          _Float16* yg16) {
  int lane = threadIdx.x & 31;
  int row = blockIdx.x * 8 + (threadIdx.x >> 5);              // b*L + l
  int b = row >> 12; int l = row & 4095;
  int hh = l >> 6, ww = l & 63;
  int lw = ww * 64 + hh;
  size_t base = (size_t)b * 4 * L_ * 64;
  float vals[2];
#pragma unroll
  for (int p = 0; p < 2; ++p) {
    int d = lane + 32 * p;
    vals[p] = oy[base + ((size_t)0 * L_ + l)           * 64 + d]
            + oy[base + ((size_t)2 * L_ + (4095 - l))  * 64 + d]
            + oy[base + ((size_t)1 * L_ + lw)          * 64 + d]
            + oy[base + ((size_t)3 * L_ + (4095 - lw)) * 64 + d];
  }
  float s = vals[0] + vals[1];
  float sq = vals[0] * vals[0] + vals[1] * vals[1];
#pragma unroll
  for (int m = 16; m >= 1; m >>= 1) { s += __shfl_xor(s, m, 32); sq += __shfl_xor(sq, m, 32); }
  float mu = s * (1.f / 64.f);
  float var = sq * (1.f / 64.f) - mu * mu;
  float rs = rsqrtf(var + EPSF);
#pragma unroll
  for (int p = 0; p < 2; ++p) {
    int d = lane + 32 * p;
    float zv = xz[(size_t)row * 128 + 64 + d];
    float nv = ((vals[p] - mu) * rs * onw[d] + onb[d]) * siluf(zv);
    yg16[(size_t)row * 64 + d] = (_Float16)nv;
  }
}

// ---------------- out_proj: (32768x64) x (64x32), accumulate into t (residual) ----------------
__global__ void k_gemm_outproj(const _Float16* A, const _Float16* Bw, float* t) {
  int lane = threadIdx.x & 31;
  int tile = blockIdx.x * 8 + (threadIdx.x >> 5);             // 4096 tiles
  int m0 = (tile >> 1) * 16, n0 = (tile & 1) * 16;
  int r15 = lane & 15; int moff = (lane >> 4) * 8;
  float* cp = t + (size_t)(m0 + moff) * 32 + n0 + r15;
  v8f c;
#pragma unroll
  for (int r = 0; r < 8; ++r) c[r] = cp[(size_t)r * 32];
  const _Float16* Arow = A + (size_t)(m0 + r15) * 64;
  const _Float16* Brow = Bw + (size_t)(n0 + r15) * 64;
#pragma unroll
  for (int s = 0; s < 2; ++s) {
    v16h a = a_frag_f16(Arow, lane, 32 * s);
    v16h b = b_frag_f16(Brow, lane, 32 * s);
    c = __builtin_amdgcn_wmma_f32_16x16x32_f16(false, a, false, b, (short)0, c, false, false);
  }
#pragma unroll
  for (int r = 0; r < 8; ++r) cp[(size_t)r * 32] = c[r];
}

// ---------------- BN + relu, NHWC t -> NCHW out ----------------
__global__ void k_writeout(const float* t, const float* g, const float* be,
                           const float* mn, const float* vr, float* out, int cbase) {
  int idx = blockIdx.x * blockDim.x + threadIdx.x;            // (b*L+l)*32 + c
  int c = idx & 31; int bl = idx >> 5; int l = bl & 4095; int b = bl >> 12;
  int ch = cbase + c;
  float v = (t[idx] - mn[ch]) * rsqrtf(vr[ch] + EPSF) * g[ch] + be[ch];
  out[(size_t)(b * 128 + ch) * L_ + l] = fmaxf(v, 0.f);
}

__global__ void k_branch4(const float* x, const float* g, const float* be,
                          const float* mn, const float* vr, float* out) {
  int idx = blockIdx.x * blockDim.x + threadIdx.x;
  int c = idx & 31; int bl = idx >> 5; int l = bl & 4095; int b = bl >> 12;
  int ch = 96 + c;
  float v = x[(size_t)(b * 128 + ch) * L_ + l];
  v = (v - mn[ch]) * rsqrtf(vr[ch] + EPSF) * g[ch] + be[ch];
  out[(size_t)(b * 128 + ch) * L_ + l] = fmaxf(v, 0.f);
}

extern "C" void kernel_launch(void* const* d_in, const int* in_sizes, int n_in,
                              void* d_out, int out_size, void* d_ws, size_t ws_size,
                              hipStream_t stream) {
  (void)in_sizes; (void)n_in; (void)out_size; (void)ws_size;
  const float* x = (const float*)d_in[0];
  const float* mh_w[3] = {(const float*)d_in[1],  (const float*)d_in[7],  (const float*)d_in[13]};
  const float* mh_b[3] = {(const float*)d_in[2],  (const float*)d_in[8],  (const float*)d_in[14]};
  const float* mw_w[3] = {(const float*)d_in[3],  (const float*)d_in[9],  (const float*)d_in[15]};
  const float* mw_b[3] = {(const float*)d_in[4],  (const float*)d_in[10], (const float*)d_in[16]};
  const float* cw3[3]  = {(const float*)d_in[5],  (const float*)d_in[11], (const float*)d_in[17]};
  const float* cb3[3]  = {(const float*)d_in[6],  (const float*)d_in[12], (const float*)d_in[18]};
  const float* ln_w = (const float*)d_in[19];
  const float* ln_b = (const float*)d_in[20];
  const float* in_proj_w = (const float*)d_in[21];
  const float* conv_w = (const float*)d_in[22];
  const float* conv_b = (const float*)d_in[23];
  const float* x_proj_w = (const float*)d_in[24];
  const float* dt_proj_w = (const float*)d_in[25];
  const float* dt_proj_b = (const float*)d_in[26];
  const float* A_log = (const float*)d_in[27];
  const float* Dp    = (const float*)d_in[28];
  const float* onw = (const float*)d_in[29];
  const float* onb = (const float*)d_in[30];
  const float* out_proj_w = (const float*)d_in[31];
  const float* bn_g = (const float*)d_in[32];
  const float* bn_b = (const float*)d_in[33];
  const float* bn_m = (const float*)d_in[34];
  const float* bn_v = (const float*)d_in[35];
  float* out = (float*)d_out;

  char* ws = (char*)d_ws;
  float*    tmpA = (float*)   (ws + 0);                        //  4 MB
  float*    tmpB = (float*)   (ws + 4194304);                  //  4 MB
  float*    t    = (float*)   (ws + 8388608);                  //  4 MB  (B,L,32) NHWC
  _Float16* ln16 = (_Float16*)(ws + 12582912);                 //  2 MB
  float*    xz   = (float*)   (ws + 14680064);                 // 16 MB  (B,L,128)
  float*    xcv  = (float*)   (ws + 31457280);                 //  8 MB  (B,L,64)
  float*    xs   = (float*)   (ws + 39845888);                 // 32 MB  (B,4,L,64), reused as out_y
  float*    xdbl = (float*)   (ws + 73400320);                 // 24 MB  (B,4,L,48)
  _Float16* yg16 = (_Float16*)(ws + 98566144);                 //  4 MB  (B,L,64)
  _Float16* w16in= (_Float16*)(ws + 102760448);
  _Float16* w16xp= (_Float16*)(ws + 102768640);
  _Float16* w16op= (_Float16*)(ws + 102793216);

  k_cvt_weights<<<48, 256, 0, stream>>>(in_proj_w, x_proj_w, out_proj_w, w16in, w16xp, w16op);
  k_branch4<<<4096, 256, 0, stream>>>(x, bn_g, bn_b, bn_m, bn_v, out);

  for (int i = 0; i < 3; ++i) {
    int cbase = 32 * i, dil = i + 1;
    k_dw_mw<<<4096, 256, 0, stream>>>(x, mw_w[i], mw_b[i], tmpA, cbase, dil);
    k_dw_mh<<<4096, 256, 0, stream>>>(tmpA, mh_w[i], mh_b[i], tmpB, dil);
    k_dw_c <<<4096, 256, 0, stream>>>(tmpB, x, cw3[i], cb3[i], t, cbase, dil);
    k_ln<<<4096, 256, 0, stream>>>(t, ln_w, ln_b, ln16);
    k_gemm_inproj<<<2048, 256, 0, stream>>>(ln16, w16in, xz);
    k_conv_silu<<<8192, 256, 0, stream>>>(xz, conv_w, conv_b, xcv);
    k_build_xs<<<8192, 256, 0, stream>>>(xcv, xs);
    k_gemm_xproj<<<3072, 256, 0, stream>>>(xs, w16xp, xdbl);
    k_scan<<<32, 64, 0, stream>>>(xdbl, xs, dt_proj_w, dt_proj_b, A_log, Dp);
    k_combine<<<4096, 256, 0, stream>>>(xs, xz, onw, onb, yg16);
    k_gemm_outproj<<<512, 256, 0, stream>>>(yg16, w16op, t);
    k_writeout<<<4096, 256, 0, stream>>>(t, bn_g, bn_b, bn_m, bn_v, out, cbase);
  }
}
